// HighwayMaxoutNetwork_5488968204798
// MI455X (gfx1250) — matrix operand order repaired and checked by reading, not compiled, not run on hardware
//
#include <hip/hip_runtime.h>
#include <hip/hip_bf16.h>
#include <stdint.h>

// ---------------------------------------------------------------------------
// HMN decoder step for MI455X (gfx1250): bf16 WMMA GEMMs with fused
// bias+maxout-pool epilogue, double-buffered TDM-staged weight tiles in LDS.
// ---------------------------------------------------------------------------

typedef __bf16 bf16_t;
typedef __attribute__((ext_vector_type(16))) __bf16    v16bf;
typedef __attribute__((ext_vector_type(8)))  float     v8f;
typedef __attribute__((ext_vector_type(4)))  unsigned  ux4;
typedef __attribute__((ext_vector_type(4)))  int       ix4;
typedef __attribute__((ext_vector_type(8)))  int       ix8;

union Frag16 { v16bf v; ux4 q[2]; };

#define D_MODEL 512
#define BATCH   32
#define SEQ     600
#define MROWS   (BATCH * SEQ)          // 19200

#if defined(__HIP_DEVICE_COMPILE__) && \
    __has_builtin(__builtin_amdgcn_tensor_load_to_lds) && \
    __has_builtin(__builtin_amdgcn_s_wait_tensorcnt)
#define HMN_USE_TDM 1
#else
#define HMN_USE_TDM 0
#endif

__device__ __forceinline__ float sigf(float x) { return 1.0f / (1.0f + __expf(-x)); }

// ---------------------------------------------------------------------------
// r = tanh([hidden; start; end] @ Wd + bd)  -> bf16     (grid 32, block 512)
// ---------------------------------------------------------------------------
__global__ void k_r(const float* __restrict__ hid, const float* __restrict__ ss,
                    const float* __restrict__ es, const float* __restrict__ Wd,
                    const float* __restrict__ bd, bf16_t* __restrict__ rbf) {
  const int b = blockIdx.x, j = threadIdx.x;
  float acc = bd[j];
  for (int k = 0; k < 512; ++k)  acc += hid[b * 512 + k] * Wd[(size_t)k * 512 + j];
  for (int k = 0; k < 1024; ++k) acc += ss[b * 1024 + k] * Wd[(size_t)(512 + k) * 512 + j];
  for (int k = 0; k < 1024; ++k) acc += es[b * 1024 + k] * Wd[(size_t)(1536 + k) * 512 + j];
  rbf[b * 512 + j] = (bf16_t)tanhf(acc);
}

// ---------------------------------------------------------------------------
// Wt[n*K + k] = (bf16) W[k*N + n]   (cast + transpose for B-fragment loads)
// ---------------------------------------------------------------------------
__global__ void k_cast_t(const float* __restrict__ W, bf16_t* __restrict__ Wt,
                         int K, int N) {
  unsigned t = blockIdx.x * 256u + threadIdx.x;
  if (t >= (unsigned)K * (unsigned)N) return;
  unsigned n = t / (unsigned)K, k = t % (unsigned)K;
  Wt[t] = (bf16_t)W[(size_t)k * N + n];
}

// ---------------------------------------------------------------------------
// e_r[row, 0:1024] = ut[row]; e_r[row, 1024:1536] = r[row/600]   (bf16)
// ---------------------------------------------------------------------------
__global__ void k_build_er(const float* __restrict__ ut, const bf16_t* __restrict__ rbf,
                           bf16_t* __restrict__ er) {
  size_t t = (size_t)blockIdx.x * 256 + threadIdx.x;
  if (t >= (size_t)MROWS * 1536) return;
  int row = (int)(t / 1536), col = (int)(t % 1536);
  float v = (col < 1024) ? ut[(size_t)row * 1024 + col]
                         : (float)rbf[(row / SEQ) * 512 + (col - 1024)];
  er[t] = (bf16_t)v;
}

// ---------------------------------------------------------------------------
// Stage a 64(n) x 64(k) bf16 weight tile into LDS.
// TDM path: wave 0 issues tensor_load_to_lds (async, TENSORcnt-tracked).
// ---------------------------------------------------------------------------
__device__ __forceinline__ void stage_tile(const bf16_t* __restrict__ Wt, int K,
                                           int nBase, int k0, bf16_t* dst, int tid) {
#if HMN_USE_TDM
  if (tid < 32) {
    unsigned long long ga =
        (unsigned long long)(uintptr_t)(Wt + (size_t)nBase * K + k0);
    ux4 g0 = { 1u,                                   // count=1, user mode
               (unsigned)(uintptr_t)dst,             // lds_addr
               (unsigned)ga,                         // global_addr[31:0]
               (unsigned)((ga >> 32) & 0x01FFFFFFu) | (2u << 30) }; // type=2
    ix8 g1 = { 0x10000,                              // data_size=1 (2B)
               (int)((K & 0xFFFF) << 16),            // tensor_dim0[15:0]
               (int)(((unsigned)K >> 16) | (1024u << 16)), // td0 hi | td1 lo
               (int)(64 << 16),                      // td1 hi(0) | tile_dim0=64
               64,                                   // tile_dim1=64
               K,                                    // tensor_dim0_stride lo
               0, 0 };
    ix4 gz = { 0, 0, 0, 0 };
#if __clang_major__ >= 23
    ix8 gz8 = { 0, 0, 0, 0, 0, 0, 0, 0 };
    __builtin_amdgcn_tensor_load_to_lds(g0, g1, gz, gz, gz8, 0);
#else
    __builtin_amdgcn_tensor_load_to_lds(g0, g1, gz, gz, 0);
#endif
  }
#else
  for (int i = tid; i < 64 * 64; i += 256)
    dst[i] = Wt[(size_t)(nBase + (i >> 6)) * K + k0 + (i & 63)];
#endif
}

__device__ __forceinline__ void wait_stage(int tid) {
#if HMN_USE_TDM
  if (tid < 32) __builtin_amdgcn_s_wait_tensorcnt(0);
#endif
}

// ---------------------------------------------------------------------------
// out[M, N/2] = maxpool2( A[M,K](bf16) @ Wt[N,K]^T (bf16) + bias[N] )
// block = 256 thr = 8 waves; tile 128(M) x 64(N); K in 64-deep double-buffered
// LDS stages fetched by the Tensor Data Mover, overlapped with WMMA compute.
// ---------------------------------------------------------------------------
__global__ __launch_bounds__(256) void k_gemm_maxout(
    const bf16_t* __restrict__ A, const bf16_t* __restrict__ Wt,
    const float* __restrict__ bias, bf16_t* __restrict__ out, int K, int N) {
  __shared__ __align__(16) bf16_t lbuf[2][64 * 64];  // [buf][n_local][k_local]
  const int tid   = threadIdx.x;
  const int lane  = tid & 31;
  const int wave  = tid >> 5;
  const int h     = lane >> 4;       // lane half (WMMA layout)
  const int ml    = lane & 15;
  const int mBase = blockIdx.x * 128 + wave * 16;
  const int nBase = blockIdx.y * 64;

  v8f c[4] = {};                     // 4 N-tiles of 16x16 f32 accumulators
  const bf16_t* arow = A + (size_t)(mBase + ml) * K;

  // Prologue: stage tile 0 into buffer 0
  stage_tile(Wt, K, nBase, 0, lbuf[0], tid);
  wait_stage(tid);
  __syncthreads();

  int cur = 0;
  for (int k0 = 0; k0 < K; k0 += 64) {
    const bool more = (k0 + 64) < K;
    // Kick off the next tile's DMA into the other buffer (overlaps compute)
    if (more) stage_tile(Wt, K, nBase, k0 + 64, lbuf[cur ^ 1], tid);

    __builtin_prefetch(arow + k0 + 64, 0, 1);   // -> global_prefetch_b8

    // A fragments for both K-halves (two b128 loads each, WMMA A layout)
    Frag16 a0, a1;
    a0.q[0] = *(const ux4*)(arow + k0 + h * 8);
    a0.q[1] = *(const ux4*)(arow + k0 + 16 + h * 8);
    a1.q[0] = *(const ux4*)(arow + k0 + 32 + h * 8);
    a1.q[1] = *(const ux4*)(arow + k0 + 48 + h * 8);

    // All B fragments up-front so the ds_loads pipeline and the WMMAs issue
    // back-to-back.
    Frag16 b0[4], b1[4];
    const bf16_t* lb = lbuf[cur];
#pragma unroll
    for (int t = 0; t < 4; ++t) {
      const bf16_t* bp = lb + (size_t)(t * 16 + ml) * 64 + h * 8;
      b0[t].q[0] = *(const ux4*)bp;
      b0[t].q[1] = *(const ux4*)(bp + 16);
      b1[t].q[0] = *(const ux4*)(bp + 32);
      b1[t].q[1] = *(const ux4*)(bp + 48);
    }
#pragma unroll
    for (int t = 0; t < 4; ++t)
      c[t] = __builtin_amdgcn_wmma_f32_16x16x32_bf16(
          false, a0.v, false, b0[t].v, (short)0, c[t], false, false);
#pragma unroll
    for (int t = 0; t < 4; ++t)
      c[t] = __builtin_amdgcn_wmma_f32_16x16x32_bf16(
          false, a1.v, false, b1[t].v, (short)0, c[t], false, false);

    __syncthreads();                 // all waves done reading lbuf[cur]
    if (more) wait_stage(tid);       // wave0: next tile landed
    __syncthreads();                 // release everyone onto the new buffer
    cur ^= 1;
  }

  // Epilogue: +bias, maxout over adjacent column pairs (shfl_xor 1), store bf16
#pragma unroll
  for (int t = 0; t < 4; ++t) {
    const int col = nBase + t * 16 + ml;
    const float bb = bias[col];
#pragma unroll
    for (int r = 0; r < 8; ++r) {
      float v = c[t][r] + bb;
      float pooled = fmaxf(v, __shfl_xor(v, 1, 32));
      if ((lane & 1) == 0) {
        int row = mBase + h * 8 + r;
        out[(size_t)row * (N >> 1) + (col >> 1)] = (bf16_t)pooled;
      }
    }
  }
}

// ---------------------------------------------------------------------------
// alpha[row] = max( [m1;m2] . w[:,0] + b0 , [m1;m2] . w[:,1] + b1 )
// one wave per row
// ---------------------------------------------------------------------------
__global__ void k_alpha(const bf16_t* __restrict__ m1, const bf16_t* __restrict__ m2,
                        const float* __restrict__ w2, const float* __restrict__ b2,
                        float* __restrict__ alpha, int M) {
  int wid = (int)((blockIdx.x * 256u + threadIdx.x) >> 5);
  int lane = threadIdx.x & 31;
  if (wid >= M) return;
  const bf16_t* r1 = m1 + (size_t)wid * 512;
  const bf16_t* r2 = m2 + (size_t)wid * 512;
  float a0 = 0.f, a1 = 0.f;
  for (int i = lane; i < 512; i += 32) {
    float x1 = (float)r1[i];
    a0 += x1 * w2[2 * i];
    a1 += x1 * w2[2 * i + 1];
    float x2 = (float)r2[i];
    a0 += x2 * w2[2 * (i + 512)];
    a1 += x2 * w2[2 * (i + 512) + 1];
  }
  for (int o = 16; o; o >>= 1) {
    a0 += __shfl_xor(a0, o, 32);
    a1 += __shfl_xor(a1, o, 32);
  }
  if (lane == 0) alpha[wid] = fmaxf(a0 + b2[0], a1 + b2[1]);
}

// ---------------------------------------------------------------------------
// per-batch argmax over L (+gather selected ut row); first-max tie-break
// ---------------------------------------------------------------------------
__global__ void k_argmax(const float* __restrict__ alpha, const float* __restrict__ ut,
                         float* __restrict__ state_out, float* __restrict__ idx_out,
                         int L) {
  const int b = blockIdx.x;
  __shared__ float sv[256];
  __shared__ int   si[256];
  float best = -1e30f;
  int   bi   = 0;
  for (int i = threadIdx.x; i < L; i += 256) {
    float v = alpha[b * L + i];
    if (v > best || (v == best && i < bi)) { best = v; bi = i; }
  }
  sv[threadIdx.x] = best; si[threadIdx.x] = bi;
  __syncthreads();
  for (int s = 128; s; s >>= 1) {
    if ((int)threadIdx.x < s) {
      float ov = sv[threadIdx.x + s]; int oi = si[threadIdx.x + s];
      if (ov > sv[threadIdx.x] || (ov == sv[threadIdx.x] && oi < si[threadIdx.x])) {
        sv[threadIdx.x] = ov; si[threadIdx.x] = oi;
      }
    }
    __syncthreads();
  }
  const int idx = si[0];
  if (threadIdx.x == 0) ((int*)idx_out)[b] = idx;   // int32 bit pattern
  for (int j = threadIdx.x; j < 1024; j += 256)
    state_out[b * 1024 + j] = ut[((size_t)b * SEQ + idx) * 1024 + j];
}

// ---------------------------------------------------------------------------
// LSTM gates = [ss;es]@wih + bih + hidden@whh + bhh      (32 x 2048)
// ---------------------------------------------------------------------------
__global__ void k_gates(const float* __restrict__ ss, const float* __restrict__ es,
                        const float* __restrict__ hid, const float* __restrict__ wih,
                        const float* __restrict__ bih, const float* __restrict__ whh,
                        const float* __restrict__ bhh, float* __restrict__ gates) {
  int t = blockIdx.x * 256 + threadIdx.x;       // 0..65535
  int b = t >> 11, j = t & 2047;
  float acc = bih[j] + bhh[j];
  const float* x1 = ss + b * 1024;
  const float* x2 = es + b * 1024;
  for (int k = 0; k < 1024; ++k) acc += x1[k] * wih[(size_t)k * 2048 + j];
  for (int k = 0; k < 1024; ++k) acc += x2[k] * wih[(size_t)(1024 + k) * 2048 + j];
  const float* hh = hid + b * 512;
  for (int k = 0; k < 512; ++k) acc += hh[k] * whh[(size_t)k * 2048 + j];
  gates[t] = acc;
}

__global__ void k_cellh(const float* __restrict__ gates, const float* __restrict__ cell_in,
                        float* __restrict__ cell_out, float* __restrict__ hvec) {
  int t = blockIdx.x * 256 + threadIdx.x;       // 0..16383
  int b = t >> 9, j = t & 511;
  const float* g = gates + b * 2048;
  float gi = sigf(g[j]), gf = sigf(g[512 + j]);
  float gg = tanhf(g[1024 + j]), go = sigf(g[1536 + j]);
  float cn = gf * cell_in[t] + gi * gg;
  cell_out[t] = cn;
  hvec[t] = go * tanhf(cn);
}

__global__ void k_mlp(const float* __restrict__ hvec, const float* __restrict__ w,
                      const float* __restrict__ bb, float* __restrict__ hid_out) {
  int t = blockIdx.x * 256 + threadIdx.x;       // 0..16383
  int b = t >> 9, j = t & 511;
  float acc = bb[j];
  const float* hv = hvec + b * 512;
  for (int k = 0; k < 512; ++k) acc += hv[k] * w[(size_t)k * 512 + j];
  hid_out[t] = acc;
}

// ---------------------------------------------------------------------------
extern "C" void kernel_launch(void* const* d_in, const int* in_sizes, int n_in,
                              void* d_out, int out_size, void* d_ws, size_t ws_size,
                              hipStream_t stream) {
  (void)in_sizes; (void)n_in; (void)out_size; (void)ws_size;

  const float* ut   = (const float*)d_in[0];
  const float* ss0  = (const float*)d_in[1];
  const float* es0  = (const float*)d_in[2];
  const float* hid0 = (const float*)d_in[3];
  const float* cel0 = (const float*)d_in[4];
  const float* Wd   = (const float*)d_in[5];
  const float* bd   = (const float*)d_in[6];
  const float* ms0w = (const float*)d_in[7];
  const float* ms0b = (const float*)d_in[8];
  const float* ms1w = (const float*)d_in[9];
  const float* ms1b = (const float*)d_in[10];
  const float* ms2w = (const float*)d_in[11];
  const float* ms2b = (const float*)d_in[12];
  const float* me0w = (const float*)d_in[13];
  const float* me0b = (const float*)d_in[14];
  const float* me1w = (const float*)d_in[15];
  const float* me1b = (const float*)d_in[16];
  const float* me2w = (const float*)d_in[17];
  const float* me2b = (const float*)d_in[18];
  const float* wih  = (const float*)d_in[19];
  const float* bih  = (const float*)d_in[20];
  const float* whh  = (const float*)d_in[21];
  const float* bhh  = (const float*)d_in[22];
  const float* mlpw = (const float*)d_in[23];
  const float* mlpb = (const float*)d_in[24];

  float* out     = (float*)d_out;
  float* o_alpha = out;                 // 19200
  float* o_beta  = out + 19200;         // 19200
  float* o_hid   = out + 38400;         // 16384
  float* o_cell  = out + 54784;         // 16384
  float* o_ss    = out + 71168;         // 32768
  float* o_es    = out + 103936;        // 32768
  float* o_start = out + 136704;        // 32 (int32 bits)
  float* o_end   = out + 136736;        // 32 (int32 bits)

  char* ws = (char*)d_ws;
  size_t off = 0;
  auto take = [&](size_t bytes) -> char* {
    char* p = ws + off;
    off += (bytes + 255) & ~(size_t)255;
    return p;
  };
  bf16_t* rbf = (bf16_t*)take((size_t)32 * 512 * 2);
  bf16_t* er  = (bf16_t*)take((size_t)MROWS * 1536 * 2);
  bf16_t* wm0 = (bf16_t*)take((size_t)1024 * 1536 * 2);
  bf16_t* wm1 = (bf16_t*)take((size_t)1024 * 512 * 2);
  bf16_t* wn0 = (bf16_t*)take((size_t)1024 * 1536 * 2);
  bf16_t* wn1 = (bf16_t*)take((size_t)1024 * 512 * 2);
  bf16_t* m1  = (bf16_t*)take((size_t)MROWS * 512 * 2);
  bf16_t* m2  = (bf16_t*)take((size_t)MROWS * 512 * 2);
  bf16_t* n1  = (bf16_t*)take((size_t)MROWS * 512 * 2);
  bf16_t* n2  = (bf16_t*)take((size_t)MROWS * 512 * 2);
  float* gts  = (float*)take((size_t)32 * 2048 * 4);
  float* hvec = (float*)take((size_t)32 * 512 * 4);

  // r projection + bf16 staging
  k_r<<<dim3(32), dim3(512), 0, stream>>>(hid0, ss0, es0, Wd, bd, rbf);
  k_cast_t<<<dim3(6144), dim3(256), 0, stream>>>(ms0w, wm0, 1536, 1024);
  k_cast_t<<<dim3(2048), dim3(256), 0, stream>>>(ms1w, wm1, 512, 1024);
  k_cast_t<<<dim3(6144), dim3(256), 0, stream>>>(me0w, wn0, 1536, 1024);
  k_cast_t<<<dim3(2048), dim3(256), 0, stream>>>(me1w, wn1, 512, 1024);
  k_build_er<<<dim3(115200), dim3(256), 0, stream>>>(ut, rbf, er);

  // WMMA maxout GEMMs (M=19200): grid = (19200/128, 1024/64)
  dim3 gg(150, 16), gb(256);
  k_gemm_maxout<<<gg, gb, 0, stream>>>(er, wm0, ms0b, m1, 1536, 1024);
  k_gemm_maxout<<<gg, gb, 0, stream>>>(er, wn0, me0b, n1, 1536, 1024);
  k_gemm_maxout<<<gg, gb, 0, stream>>>(m1, wm1, ms1b, m2, 512, 1024);
  k_gemm_maxout<<<gg, gb, 0, stream>>>(n1, wn1, me1b, n2, 512, 1024);

  // scores + argmax + gather
  k_alpha<<<dim3(2400), dim3(256), 0, stream>>>(m1, m2, ms2w, ms2b, o_alpha, MROWS);
  k_alpha<<<dim3(2400), dim3(256), 0, stream>>>(n1, n2, me2w, me2b, o_beta, MROWS);
  k_argmax<<<dim3(32), dim3(256), 0, stream>>>(o_alpha, ut, o_ss, o_start, SEQ);
  k_argmax<<<dim3(32), dim3(256), 0, stream>>>(o_beta, ut, o_es, o_end, SEQ);

  // LSTM cell + MLP
  k_gates<<<dim3(256), dim3(256), 0, stream>>>(o_ss, o_es, hid0, wih, bih, whh, bhh, gts);
  k_cellh<<<dim3(64), dim3(256), 0, stream>>>(gts, cel0, o_cell, hvec);
  k_mlp<<<dim3(64), dim3(256), 0, stream>>>(hvec, mlpw, mlpb, o_hid);
}
